// DecoderLayer_39195871543516
// MI455X (gfx1250) — compile-verified, matrix operand order checked by reading, not run on hardware
//
#include <hip/hip_runtime.h>
#include <hip/hip_bf16.h>

// ---------------------------------------------------------------------------
// Decoder layer (B=8, S=512, D=1024, H=16, HD=64, DFF=4096) for gfx1250.
// All GEMMs via v_wmma_f32_16x16x32_bf16 (wave32 WMMA), bf16 data / f32 accum.
// GEMM: 8 waves/block, each wave a 32x64 tile (2 A-frags x 4 B-frags,
// 8 accumulators) for ~22 FLOP/B cache-side intensity + deep XDL pipelining.
// ---------------------------------------------------------------------------

typedef __attribute__((ext_vector_type(16))) __bf16 v16bf;
typedef __attribute__((ext_vector_type(8)))  __bf16 v8bf;
typedef __attribute__((ext_vector_type(8)))  float  v8f;

#define BATCH 8
#define SEQ   512
#define DIM   1024
#define HEADS 16
#define HDIM  64
#define DFF   4096
#define ROWS  (BATCH * SEQ)          // 4096
#define LN_EPS 1e-3f

// ---- WMMA helpers ----------------------------------------------------------

__device__ __forceinline__ v8f wmma_bf16(v16bf a, v16bf b, v8f c) {
  // 8 args: (neg_a, A, neg_b, B, c_mod, C, reuse_a, reuse_b)
  return __builtin_amdgcn_wmma_f32_16x16x32_bf16(false, a, false, b, (short)0, c,
                                                 false, false);
}

// Load one 16x32 bf16 fragment (A layout; B is loaded as B^T rows, same layout).
// Per ISA 7.12.2: lanes 0-15 hold row idx, K {0..7,16..23}; lanes 16-31 same
// rows, K {8..15,24..31}.  base points at (row0, k=0); ld = row stride (elems).
__device__ __forceinline__ v16bf load_frag(const __bf16* __restrict__ base,
                                           int ld, int k0) {
  const int lane = threadIdx.x & 31;
  const int half = (lane >> 4) & 1;
  const int idx  = lane & 15;
  const __bf16* p = base + (size_t)idx * ld + k0 + half * 8;
  union { v16bf v; v8bf h[2]; } u;
  u.h[0] = *(const v8bf*)(p);        // K block lo (8 elems, 16B aligned)
  u.h[1] = *(const v8bf*)(p + 16);   // K block hi
  return u.v;
}

// ---- conversion / repack kernels ------------------------------------------

__global__ void k_cvt_bf16(const float* __restrict__ x, __bf16* __restrict__ o,
                           int n) {
  int i = blockIdx.x * 256 + threadIdx.x;
  if (i < n) o[i] = (__bf16)x[i];
}

// wq[H][D][HD] f32  ->  out[(h*HD+d)][f] bf16   (N=1024 rows of K=1024)
__global__ void k_pack_headw(const float* __restrict__ w,
                             __bf16* __restrict__ out) {
  int i = blockIdx.x * 256 + threadIdx.x;          // 0 .. 1M-1
  int n = i >> 10, f = i & 1023;
  int h = n >> 6,  d = n & 63;
  out[i] = (__bf16)w[((size_t)h * DIM + f) * HDIM + d];
}

// w[Kdim][Ndim] f32 -> out[Ndim][Kdim] bf16
__global__ void k_transpose_cvt(const float* __restrict__ w,
                                __bf16* __restrict__ out, int Kdim, int Ndim) {
  int i = blockIdx.x * 256 + threadIdx.x;
  if (i >= Kdim * Ndim) return;
  int n = i / Kdim, k = i % Kdim;
  out[i] = (__bf16)w[(size_t)k * Ndim + n];
}

// V[(b*512+s)*1024 + h*64 + d] -> Vt[((b*16+h)*64+d)*512 + s]
__global__ void k_trans_v(const __bf16* __restrict__ V,
                          __bf16* __restrict__ Vt) {
  int i = blockIdx.x * 256 + threadIdx.x;          // 0 .. 4M-1
  int s = i & 511;
  int d = (i >> 9) & 63;
  int h = (i >> 15) & 15;
  int b = i >> 19;
  Vt[i] = V[((size_t)(b * SEQ + s)) * DIM + h * HDIM + d];
}

// ---- generic WMMA GEMM: C[M,N] = A[M,K] * Bt[N,K]^T ------------------------
// 8 waves/block in a 4(M) x 2(N) grid; each wave: 32(M) x 64(N) tile.
// Block tile 128x128.

template <int BIAS, int RELU>
__global__ void k_gemm(const __bf16* __restrict__ A,
                       const __bf16* __restrict__ Bt,
                       const float* __restrict__ bias,
                       float* __restrict__ outF,
                       __bf16* __restrict__ outB,
                       int M, int N, int K) {
  const int lane = threadIdx.x & 31;
  const int wave = threadIdx.x >> 5;               // 0..7
  const int wm = wave & 3, wn = wave >> 2;         // 4 x 2 wave grid
  const int mBase = blockIdx.x * 128 + wm * 32;
  const int nBase = blockIdx.y * 128 + wn * 64;

  v8f acc[2][4] = {};
  const __bf16* aBase = A + (size_t)mBase * K;
  for (int k0 = 0; k0 < K; k0 += 32) {
    // prefetch next K step of the two A row-blocks (global_prefetch_b8)
    if (k0 + 32 < K) {
      __builtin_prefetch(aBase + k0 + 32, 0, 3);
      __builtin_prefetch(aBase + (size_t)16 * K + k0 + 32, 0, 3);
    }
    v16bf a0 = load_frag(aBase, K, k0);
    v16bf a1 = load_frag(aBase + (size_t)16 * K, K, k0);
#pragma unroll
    for (int j = 0; j < 4; ++j) {
      v16bf b = load_frag(Bt + (size_t)(nBase + 16 * j) * K, K, k0);
      acc[0][j] = wmma_bf16(a0, b, acc[0][j]);
      acc[1][j] = wmma_bf16(a1, b, acc[1][j]);
    }
  }

  const int half = (lane >> 4) & 1, idx = lane & 15;
#pragma unroll
  for (int i = 0; i < 2; ++i) {
#pragma unroll
    for (int j = 0; j < 4; ++j) {
#pragma unroll
      for (int v = 0; v < 8; ++v) {
        int r = mBase + 16 * i + v + 8 * half;     // C/D layout (ISA 7.12.2)
        int c = nBase + 16 * j + idx;
        float val = acc[i][j][v];
        if (BIAS) val += bias[c];
        if (RELU) val = fmaxf(val, 0.0f);
        size_t o = (size_t)r * N + c;
        if (outF) outF[o] = val;
        if (outB) outB[o] = (__bf16)val;
      }
    }
  }
}

// ---- fused attention: scores -> mask -> softmax -> P@V ---------------------
// One wave (blockDim=32) handles 16 query rows of one (b,h).
// maskMode 0: causal mask[(q)*512+s] ; maskMode 1: pad mask[b*512+s].

__global__ void k_attn(const __bf16* __restrict__ Q,
                       const __bf16* __restrict__ Kact,
                       const __bf16* __restrict__ Vt,
                       const float* __restrict__ mask, int maskMode,
                       float scale, __bf16* __restrict__ ctx) {
  __shared__ float  Sf[16 * SEQ];   // 32 KB
  __shared__ __bf16 Pb[16 * SEQ];   // 16 KB

  const int lane = threadIdx.x & 31;
  const int qt = blockIdx.x & 31;
  const int h  = (blockIdx.x >> 5) & 15;
  const int b  = blockIdx.x >> 9;
  const int q0 = qt * 16;
  const int half = lane >> 4, idx = lane & 15;

  const __bf16* qBase = Q + (size_t)(b * SEQ + q0) * DIM + h * HDIM;

  // scores: 32 key-tiles of 16, K-loop over HDIM=64 (2 WMMAs)
  for (int s0 = 0; s0 < SEQ; s0 += 16) {
    const __bf16* kBase = Kact + (size_t)(b * SEQ + s0) * DIM + h * HDIM;
    v8f acc = {};
    acc = wmma_bf16(load_frag(qBase, DIM, 0),  load_frag(kBase, DIM, 0),  acc);
    acc = wmma_bf16(load_frag(qBase, DIM, 32), load_frag(kBase, DIM, 32), acc);
#pragma unroll
    for (int v = 0; v < 8; ++v) {
      int r = v + 8 * half;
      int c = s0 + idx;
      float mv = (maskMode == 0) ? mask[(size_t)(q0 + r) * SEQ + c]
                                 : mask[(size_t)b * SEQ + c];
      Sf[r * SEQ + c] = acc[v] * scale + mv * (-1e9f);
    }
  }
  __syncthreads();

  // softmax (lanes 0-15, one row each)
  if (lane < 16) {
    const int r = lane;
    float mx = -3.4e38f;
    for (int s = 0; s < SEQ; ++s) mx = fmaxf(mx, Sf[r * SEQ + s]);
    float sum = 0.0f;
    for (int s = 0; s < SEQ; ++s) {
      float e = __expf(Sf[r * SEQ + s] - mx);
      Sf[r * SEQ + s] = e;
      sum += e;
    }
    float inv = 1.0f / sum;
    for (int s = 0; s < SEQ; ++s)
      Pb[r * SEQ + s] = (__bf16)(Sf[r * SEQ + s] * inv);
  }
  __syncthreads();

  // O = P[16,512] @ V[512,64]  (Bt = per-head transposed V [64][512])
  const __bf16* vtBase = Vt + (size_t)(b * HEADS + h) * HDIM * SEQ;
  v8f oacc[4] = {};
  for (int k0 = 0; k0 < SEQ; k0 += 32) {
    v16bf a = load_frag((const __bf16*)Pb, SEQ, k0);
#pragma unroll
    for (int j = 0; j < 4; ++j)
      oacc[j] = wmma_bf16(a, load_frag(vtBase + (size_t)(16 * j) * SEQ, SEQ, k0),
                          oacc[j]);
  }
#pragma unroll
  for (int j = 0; j < 4; ++j)
#pragma unroll
    for (int v = 0; v < 8; ++v) {
      int r = v + 8 * half;
      int c = 16 * j + idx;
      ctx[(size_t)(b * SEQ + q0 + r) * DIM + h * HDIM + c] = (__bf16)oacc[j][v];
    }
}

// ---- residual + LayerNorm (one 256-thread block per row of D=1024) ---------

__global__ void k_add_ln(const float* __restrict__ x,
                         const float* __restrict__ res,
                         const float* __restrict__ g,
                         const float* __restrict__ bta,
                         float* __restrict__ outF,
                         __bf16* __restrict__ outB) {
  __shared__ float buf[256];
  const int t = threadIdx.x;
  const size_t off = (size_t)blockIdx.x * DIM;

  float v[4], s = 0.0f;
#pragma unroll
  for (int i = 0; i < 4; ++i) {
    v[i] = x[off + t + i * 256] + res[off + t + i * 256];
    s += v[i];
  }
  buf[t] = s; __syncthreads();
  for (int k = 128; k > 0; k >>= 1) { if (t < k) buf[t] += buf[t + k]; __syncthreads(); }
  const float mean = buf[0] * (1.0f / DIM);
  __syncthreads();

  float s2 = 0.0f;
#pragma unroll
  for (int i = 0; i < 4; ++i) { float d = v[i] - mean; s2 += d * d; }
  buf[t] = s2; __syncthreads();
  for (int k = 128; k > 0; k >>= 1) { if (t < k) buf[t] += buf[t + k]; __syncthreads(); }
  const float inv = rsqrtf(buf[0] * (1.0f / DIM) + LN_EPS);

#pragma unroll
  for (int i = 0; i < 4; ++i) {
    int c = t + i * 256;
    float y = g[c] * (v[i] - mean) * inv + bta[c];
    outF[off + c] = y;
    if (outB) outB[off + c] = (__bf16)y;
  }
}

// ---------------------------------------------------------------------------

extern "C" void kernel_launch(void* const* d_in, const int* in_sizes, int n_in,
                              void* d_out, int out_size, void* d_ws, size_t ws_size,
                              hipStream_t stream) {
  const float* xz    = (const float*)d_in[0];
  const float* yz    = (const float*)d_in[1];
  const float* wq_s  = (const float*)d_in[2];
  const float* wk_s  = (const float*)d_in[3];
  const float* wv_s  = (const float*)d_in[4];
  const float* wo_s  = (const float*)d_in[5];
  const float* wq_c  = (const float*)d_in[6];
  const float* wk_c  = (const float*)d_in[7];
  const float* wv_c  = (const float*)d_in[8];
  const float* wo_c  = (const float*)d_in[9];
  const float* f_w1  = (const float*)d_in[10];
  const float* f_b1  = (const float*)d_in[11];
  const float* f_w2  = (const float*)d_in[12];
  const float* f_b2  = (const float*)d_in[13];
  const float* ln0g  = (const float*)d_in[14];
  const float* ln0b  = (const float*)d_in[15];
  const float* ln1g  = (const float*)d_in[16];
  const float* ln1b  = (const float*)d_in[17];
  const float* ln2g  = (const float*)d_in[18];
  const float* ln2b  = (const float*)d_in[19];
  const float* lam   = (const float*)d_in[20];   // [512,512] causal
  const float* pmask = (const float*)d_in[21];   // [B,512]

  char* ws = (char*)d_ws;
  const size_t MBy = 1ull << 20;
  __bf16* yzb  = (__bf16*)(ws + 0 * MBy);     // 8 MB
  __bf16* xzb  = (__bf16*)(ws + 8 * MBy);     // 8 MB
  __bf16* wTq  = (__bf16*)(ws + 16 * MBy);    // 2 MB
  __bf16* wTk  = (__bf16*)(ws + 18 * MBy);    // 2 MB
  __bf16* wTv  = (__bf16*)(ws + 20 * MBy);    // 2 MB
  __bf16* wTo  = (__bf16*)(ws + 22 * MBy);    // 2 MB
  __bf16* w1T  = (__bf16*)(ws + 24 * MBy);    // 8 MB
  __bf16* w2T  = (__bf16*)(ws + 32 * MBy);    // 8 MB
  __bf16* Qb   = (__bf16*)(ws + 40 * MBy);    // 8 MB
  __bf16* Kb   = (__bf16*)(ws + 48 * MBy);    // 8 MB
  __bf16* Vb   = (__bf16*)(ws + 56 * MBy);    // 8 MB
  __bf16* Vt   = (__bf16*)(ws + 64 * MBy);    // 8 MB
  __bf16* ctxb = (__bf16*)(ws + 72 * MBy);    // 8 MB
  float*  projF= (float*) (ws + 80 * MBy);    // 16 MB
  float*  o1F  = (float*) (ws + 96 * MBy);    // 16 MB
  float*  o2F  = (float*) (ws + 112 * MBy);   // 16 MB
  __bf16* o1B  = (__bf16*)(ws + 128 * MBy);   // 8 MB
  __bf16* o2B  = (__bf16*)(ws + 136 * MBy);   // 8 MB
  __bf16* ffnH = (__bf16*)(ws + 144 * MBy);   // 32 MB

  const int ACT = ROWS * DIM;                  // 4M elems
  const float scale = 1.0f / (sqrtf((float)DIM) + 1e-8f);

  dim3 blk256(256), blk32(32);
  dim3 gCvt((ACT + 255) / 256);
  dim3 gW1M((DIM * DIM + 255) / 256);
  dim3 gGemmD(ROWS / 128, DIM / 128);          // 32 x 8
  dim3 gGemmF(ROWS / 128, DFF / 128);          // 32 x 32
  dim3 gAttn(BATCH * HEADS * (SEQ / 16));
  dim3 gLN(ROWS);

  // stage activations to bf16
  k_cvt_bf16<<<gCvt, blk256, 0, stream>>>(yz, yzb, ACT);
  k_cvt_bf16<<<gCvt, blk256, 0, stream>>>(xz, xzb, ACT);

  // ===================== self attention =====================
  k_pack_headw<<<gW1M, blk256, 0, stream>>>(wq_s, wTq);
  k_pack_headw<<<gW1M, blk256, 0, stream>>>(wk_s, wTk);
  k_pack_headw<<<gW1M, blk256, 0, stream>>>(wv_s, wTv);
  k_transpose_cvt<<<gW1M, blk256, 0, stream>>>(wo_s, wTo, DIM, DIM);

  k_gemm<0,0><<<gGemmD, blk256, 0, stream>>>(yzb, wTq, nullptr, nullptr, Qb, ROWS, DIM, DIM);
  k_gemm<0,0><<<gGemmD, blk256, 0, stream>>>(yzb, wTk, nullptr, nullptr, Kb, ROWS, DIM, DIM);
  k_gemm<0,0><<<gGemmD, blk256, 0, stream>>>(yzb, wTv, nullptr, nullptr, Vb, ROWS, DIM, DIM);
  k_trans_v<<<gCvt, blk256, 0, stream>>>(Vb, Vt);

  k_attn<<<gAttn, blk32, 0, stream>>>(Qb, Kb, Vt, lam, 0, scale, ctxb);

  k_gemm<0,0><<<gGemmD, blk256, 0, stream>>>(ctxb, wTo, nullptr, projF, nullptr, ROWS, DIM, DIM);
  k_add_ln<<<gLN, blk256, 0, stream>>>(projF, yz, ln0g, ln0b, o1F, o1B);

  // ===================== cross attention =====================
  k_pack_headw<<<gW1M, blk256, 0, stream>>>(wq_c, wTq);
  k_pack_headw<<<gW1M, blk256, 0, stream>>>(wk_c, wTk);
  k_pack_headw<<<gW1M, blk256, 0, stream>>>(wv_c, wTv);
  k_transpose_cvt<<<gW1M, blk256, 0, stream>>>(wo_c, wTo, DIM, DIM);

  k_gemm<0,0><<<gGemmD, blk256, 0, stream>>>(o1B, wTq, nullptr, nullptr, Qb, ROWS, DIM, DIM);
  k_gemm<0,0><<<gGemmD, blk256, 0, stream>>>(xzb, wTk, nullptr, nullptr, Kb, ROWS, DIM, DIM);
  k_gemm<0,0><<<gGemmD, blk256, 0, stream>>>(xzb, wTv, nullptr, nullptr, Vb, ROWS, DIM, DIM);
  k_trans_v<<<gCvt, blk256, 0, stream>>>(Vb, Vt);

  k_attn<<<gAttn, blk32, 0, stream>>>(Qb, Kb, Vt, pmask, 1, scale, ctxb);

  k_gemm<0,0><<<gGemmD, blk256, 0, stream>>>(ctxb, wTo, nullptr, projF, nullptr, ROWS, DIM, DIM);
  k_add_ln<<<gLN, blk256, 0, stream>>>(projF, o1F, ln1g, ln1b, o2F, o2B);

  // ===================== FFN =====================
  dim3 gTW1((DIM * DFF + 255) / 256);
  k_transpose_cvt<<<gTW1, blk256, 0, stream>>>(f_w1, w1T, DIM, DFF);   // -> [4096][1024]
  k_transpose_cvt<<<gTW1, blk256, 0, stream>>>(f_w2, w2T, DFF, DIM);   // -> [1024][4096]

  k_gemm<1,1><<<gGemmF, blk256, 0, stream>>>(o2B, w1T, f_b1, nullptr, ffnH, ROWS, DFF, DIM);
  k_gemm<1,0><<<gGemmD, blk256, 0, stream>>>(ffnH, w2T, f_b2, projF, nullptr, ROWS, DIM, DFF);

  k_add_ln<<<gLN, blk256, 0, stream>>>(projF, o2F, ln2g, ln2b, (float*)d_out, nullptr);
}
